// BLTEB_26233660244416
// MI455X (gfx1250) — matrix operand order repaired and checked by reading, not compiled, not run on hardware
//
#include <hip/hip_runtime.h>
#include <hip/hip_bf16.h>
#include <math.h>

typedef __attribute__((ext_vector_type(16))) _Float16 v16h;
typedef __attribute__((ext_vector_type(8)))  _Float16 v8h;
typedef __attribute__((ext_vector_type(4)))  _Float16 v4h;
typedef __attribute__((ext_vector_type(8)))  float    v8f;
typedef __attribute__((ext_vector_type(4)))  float    v4f;

#define HW 512
#define NPIX (HW*HW)
#define BATCH 4
#define TT 5
#define NF 16
#define NEG_SLOPE 0.2f
#define EPSV 1e-8f

// ws float layout: [0..11] bl2d[b][c]; [16..75] blFinal[b][c][d]; [80..139] diff[b][c][d]

// ---------------------------------------------------------------------------
// Kernel 1: per-batch stats on the center frame (median R, mean/std G & B)
// ---------------------------------------------------------------------------
__global__ void stats_kernel(const float* __restrict__ I, float* __restrict__ ws) {
  int b = blockIdx.x;
  int tid = threadIdx.x;
  __shared__ unsigned int hist[256];
  __shared__ double red[256];
  __shared__ unsigned int sh_prefix;
  __shared__ int sh_k;

  double means[2], stds[2];
  for (int cc = 0; cc < 2; ++cc) {
    const float* p = I + (((size_t)b*3 + (cc+1))*TT + 2) * (size_t)NPIX;
    double s = 0.0, s2 = 0.0;
    for (int idx = tid; idx < NPIX; idx += 256) {
      float v = fminf(fmaxf(p[idx], 0.f), 1.f) * 255.f;
      s += v; s2 += (double)v * (double)v;
    }
    red[tid] = s; __syncthreads();
    for (int off = 128; off > 0; off >>= 1) { if (tid < off) red[tid] += red[tid+off]; __syncthreads(); }
    double tot = red[0]; __syncthreads();
    red[tid] = s2; __syncthreads();
    for (int off = 128; off > 0; off >>= 1) { if (tid < off) red[tid] += red[tid+off]; __syncthreads(); }
    double tot2 = red[0]; __syncthreads();
    double m = tot / (double)NPIX;
    double var = tot2 / (double)NPIX - m*m;
    if (var < 0.0) var = 0.0;
    means[cc] = m; stds[cc] = sqrt(var);
  }

  // Exact median of R: MSB-first radix-select on IEEE bit patterns (vals >= 0)
  const float* pr = I + (((size_t)b*3 + 0)*TT + 2) * (size_t)NPIX;
  float medv[2];
  for (int sel = 0; sel < 2; ++sel) {
    int k = NPIX/2 - 1 + sel;
    unsigned int prefix = 0, mask = 0;
    for (int pass = 0; pass < 4; ++pass) {
      int shift = 24 - 8*pass;
      hist[tid] = 0;
      __syncthreads();
      for (int idx = tid; idx < NPIX; idx += 256) {
        float v = fminf(fmaxf(pr[idx], 0.f), 1.f) * 255.f;
        unsigned int u = __float_as_uint(v);
        if ((u & mask) == prefix) atomicAdd(&hist[(u >> shift) & 0xff], 1u);
      }
      __syncthreads();
      if (tid == 0) {
        unsigned int cum = 0; int bsel = 255;
        for (int bin = 0; bin < 256; ++bin) {
          unsigned int h = hist[bin];
          if (cum + h > (unsigned int)k) { bsel = bin; break; }
          cum += h;
        }
        sh_prefix = prefix | ((unsigned int)bsel << shift);
        sh_k = k - (int)cum;
      }
      __syncthreads();
      prefix = sh_prefix; k = sh_k;
      mask |= 0xffu << shift;
      __syncthreads();
    }
    medv[sel] = __uint_as_float(prefix);
  }

  if (tid == 0) {
    float r_med = 0.5f * (medv[0] + medv[1]);
    float bl_r = 140.f / (1.f + 14.4f * expf(-0.034f * r_med));
    float bl_g = 1.13f*(float)means[0] + 1.11f*(float)stds[0] - 25.6f;
    float bl_b = 1.13f*(float)means[1] + 1.11f*(float)stds[1] - 25.6f;
    float v[3] = {bl_r, bl_g, bl_b};
    for (int c = 0; c < 3; ++c) ws[b*3 + c] = fminf(fmaxf(v[c], 5.f), 250.f) / 255.f;
  }
}

// ---------------------------------------------------------------------------
// Kernel 2: diff[b,c,d] = mean_hw(i - j)
// ---------------------------------------------------------------------------
__global__ void diff_kernel(const float* __restrict__ I, const float* __restrict__ J,
                            float* __restrict__ ws) {
  int s = blockIdx.x;                  // b*15 + c*5 + d
  int b = s / 15, r = s % 15, c = r / 5, d = r % 5;
  const float* pi = I + (((size_t)b*3 + c)*TT + d) * (size_t)NPIX;
  const float* pj = J + ((size_t)b*3 + c) * (size_t)NPIX;
  __shared__ double red[256];
  int tid = threadIdx.x;
  double sum = 0.0;
  for (int idx = tid; idx < NPIX; idx += 256) sum += (double)pi[idx] - (double)pj[idx];
  red[tid] = sum; __syncthreads();
  for (int off = 128; off > 0; off >>= 1) { if (tid < off) red[tid] += red[tid+off]; __syncthreads(); }
  if (tid == 0) ws[80 + s] = (float)(red[0] / (double)NPIX);
}

// ---------------------------------------------------------------------------
// Kernel 3: tiny 3->16->3 MLP, then blFinal = bl2d + w1*sigmoid(...)
// ---------------------------------------------------------------------------
__global__ void mlp_kernel(const float* __restrict__ wbl1, const float* __restrict__ bbl1,
                           const float* __restrict__ wbl2, const float* __restrict__ bbl2,
                           const float* __restrict__ w1, float* __restrict__ ws) {
  int t = threadIdx.x;
  if (t >= BATCH*TT) return;
  int b = t / TT, d = t % TT;
  float df[3];
  for (int c = 0; c < 3; ++c) df[c] = ws[80 + b*15 + c*5 + d];
  float h1[NF];
  for (int o = 0; o < NF; ++o) {
    float a = bbl1[o];
    for (int c = 0; c < 3; ++c) a += wbl1[o*3 + c] * df[c];
    h1[o] = (a >= 0.f) ? a : NEG_SLOPE * a;
  }
  for (int c = 0; c < 3; ++c) {
    float a = bbl2[c];
    for (int o = 0; o < NF; ++o) a += wbl2[c*NF + o] * h1[o];
    float sg = 1.f / (1.f + expf(-a));
    ws[16 + b*15 + c*5 + d] = ws[b*3 + c] + w1[0] * sg;
  }
}

// ---------------------------------------------------------------------------
// Kernel 4: fused  bl-fill + x -> conv1(3->16)+leaky -> conv2(16->1) -> sigmoid
//
// conv1: K = tap*4 + c4 (cin pad 4), x LDS [d][h][w][c4] -> A frag = 4x ds_load_b64
//        (tap offsets are compile-time constants, hi/lo selected once per wave;
//         the single invalid hi-half tap reads a zeroed LDS slab: no branches).
//        h2 rows padded 324->336 so stores need no guards (pad rows never read).
// conv2: K = (kh*3+kw)*80 + (c*5+dd), h2 LDS [hh][ww][c*5+dd] -> A frag =
//        2x ds_load_b128 with pure immediate offsets (run offset = CONST+laneHi8).
//        N carries the 5 output depths (B holds depth-shifted weights).
// ---------------------------------------------------------------------------
#define XS_N   (7*400*4)     // 11200 f16: x [dpad(7)][20][20][c4]
#define H2_N   (336*80)      // 26880 f16: h2 [s(324 used, 336 alloc)][c*5+dd]
#define B1_N   (4*512)       //  2048 f16: conv1 B frags [kstep][lane][e]
#define B2_N   (23*512)      // 11776 f16: conv2 B frags [kstep][lane][e]
#define ZOFF   (XS_N + H2_N + B1_N + B2_N + 32)   // f16 units: zero slab (8 f16)
#define SMEM_BYTES ((ZOFF + 8) * 2)

// tap t = (kd,kh,kw): offset in x (f16 units)
#define TAPOFF(t) ((((t)/9)*1600) + ((((t)%9)/3)*80) + (((t)%3)*4))
// conv2 8-run: k0 -> offset in h2 (f16 units)
#define RUNOFFK(k0) (((((k0)/80)/3)*1440) + ((((k0)/80)%3)*80) + ((k0)%80))
// assemble v16h from four v4h
#define MK16(A,B,C,D) __builtin_shufflevector( \
    __builtin_shufflevector(A, B, 0,1,2,3,4,5,6,7), \
    __builtin_shufflevector(C, D, 0,1,2,3,4,5,6,7), \
    0,1,2,3,4,5,6,7,8,9,10,11,12,13,14,15)

__global__ void __launch_bounds__(256) fused_conv_kernel(
    const float* __restrict__ I, const float* __restrict__ J,
    const float* __restrict__ wt1, const float* __restrict__ bt1,
    const float* __restrict__ wt2, const float* __restrict__ bt2,
    const float* __restrict__ w2p, const float* __restrict__ tbias,
    const float* __restrict__ ws, float* __restrict__ blOut,
    float* __restrict__ outT)
{
  extern __shared__ __align__(32) char smem[];
  _Float16* xs   = (_Float16*)smem;         // XS_N
  _Float16* h2s  = xs  + XS_N;              // H2_N
  _Float16* B1   = h2s + H2_N;              // B1_N
  _Float16* B2   = B1  + B1_N;              // B2_N
  float* blv     = (float*)(B2 + B2_N);     // 16 floats (15 used)

  const int tid  = threadIdx.x;
  const int lane = tid & 31;
  const int wave = tid >> 5;
  const int b  = blockIdx.z;
  const int h0 = blockIdx.y * 16;
  const int w0 = blockIdx.x * 16;
  const float w2 = w2p[0];
  const int laneHi8 = (lane & 16) >> 1;     // 8 for upper half-wave
  const bool hiF = (lane & 16) != 0;

  if (tid < 15) blv[tid] = ws[16 + b*15 + tid];            // [c*5+d]
  if (tid == 0) *(v4f*)(xs + ZOFF) = (v4f){};              // zero slab (16B)
  for (int idx = tid; idx < B1_N; idx += 256) {            // conv1 B operand
    int ks = idx >> 9, l = (idx >> 4) & 31, e = idx & 15;
    int K = ks*32 + (e & 7) + ((l & 16) >> 1) + ((e & 8) << 1);
    int tap = K >> 2, c = K & 3;
    int co = l & 15;
    B1[idx] = (_Float16)((tap < 27 && c < 3) ? wt1[co*81 + c*27 + tap] : 0.f);
  }
  for (int idx = tid; idx < B2_N; idx += 256) {            // conv2 B: depth-shifted w_t2
    int ks = idx >> 9, l = (idx >> 4) & 31, e = idx & 15;
    int K = ks*32 + (e & 7) + ((l & 16) >> 1) + ((e & 8) << 1);
    int d = l & 15;
    float v = 0.f;
    if (K < 720 && d < 5) {
      int blk = K/80, rem = K%80, c = rem/5, dd = rem%5;
      int kd = dd - d + 1;
      if (kd >= 0 && kd < 3) v = wt2[c*27 + kd*9 + blk];
    }
    B2[idx] = (_Float16)v;
  }
  __syncthreads();

  // ---- bl output: broadcast-fill this block's 16x16 region (NT stores) ----
  for (int idx = tid; idx < 15*64; idx += 256) {           // 15 slices x 64 float4
    int c5d = idx >> 6, rem = idx & 63;
    int ph = rem >> 2, pw4 = (rem & 3) << 2;
    float s = blv[c5d];
    v4f v = {s, s, s, s};
    size_t o = ((size_t)(b*15 + c5d))*(size_t)NPIX + (size_t)(h0 + ph)*HW + (w0 + pw4);
    __builtin_nontemporal_store(v, (v4f*)(blOut + o));
  }

  // ---- stage x (halo 2, depth zero-pad, c padded to 4) as packed b64 ----
  for (int idx = tid; idx < 7*400; idx += 256) {
    int dp = idx / 400, s = idx % 400;
    int hh = s / 20, ww = s % 20;
    int d = dp - 1;
    int gh = h0 - 2 + hh, gw = w0 - 2 + ww;
    v4h q = {};
    if (d >= 0 && d < TT && gh >= 0 && gh < HW && gw >= 0 && gw < HW) {
      size_t sp = (size_t)gh*HW + gw;
#pragma unroll
      for (int c = 0; c < 3; ++c) {
        float iv = I[(((size_t)(b*3 + c))*TT + d)*(size_t)NPIX + sp];
        float jv = J[((size_t)(b*3 + c))*(size_t)NPIX + sp];
        float bl = blv[c*5 + d];
        q[c] = (_Float16)(iv + w2 * ((iv - bl) / (jv - bl + EPSV)));
      }
    }
    *(v4h*)(xs + idx*4) = q;
  }
  __syncthreads();

  // ---- conv1: per-depth tiles (5 x 21), N=16 channels, K=108 (pad 128) ----
  v16h b1f0 = *(const v16h*)(B1 + 0*512 + lane*16);
  v16h b1f1 = *(const v16h*)(B1 + 1*512 + lane*16);
  v16h b1f2 = *(const v16h*)(B1 + 2*512 + lane*16);
  v16h b1f3 = *(const v16h*)(B1 + 3*512 + lane*16);
  const float bias1 = bt1[lane & 15];

  // per-wave-half tap offsets (compile-time constants, selected once)
  int so[13];
#pragma unroll
  for (int ks = 0; ks < 3; ++ks) {
    so[ks*4+0] = hiF ? TAPOFF(ks*8+2) : TAPOFF(ks*8+0);
    so[ks*4+1] = hiF ? TAPOFF(ks*8+3) : TAPOFF(ks*8+1);
    so[ks*4+2] = hiF ? TAPOFF(ks*8+6) : TAPOFF(ks*8+4);
    so[ks*4+3] = hiF ? TAPOFF(ks*8+7) : TAPOFF(ks*8+5);
  }
  so[12] = hiF ? TAPOFF(26) : TAPOFF(24);
  const int so13 = hiF ? 0 : TAPOFF(25);   // hi-half tap 27 invalid -> zero slab

  for (int t = wave; t < 105; t += 8) {                    // 5 depths x 21 tiles
    int d = t / 21, stile = t - d*21;
    int s = stile*16 + (lane & 15);
    int sc = (s < 324) ? s : 323;
    int dpx = d*1600 + (sc/18)*80 + (sc%18)*4;

    v4h q[13];
#pragma unroll
    for (int sl = 0; sl < 13; ++sl) q[sl] = *(const v4h*)(xs + dpx + so[sl]);
    int a13 = dpx + so13;
    a13 = hiF ? ZOFF : a13;
    v4h q13 = *(const v4h*)(xs + a13);

    v4h zq = {};
    v16h af0 = MK16(q[0],  q[1],  q[2],  q[3]);
    v16h af1 = MK16(q[4],  q[5],  q[6],  q[7]);
    v16h af2 = MK16(q[8],  q[9],  q[10], q[11]);
    v16h af3 = MK16(q[12], q13,   zq,    zq);

    v8f acc = {};
    acc = __builtin_amdgcn_wmma_f32_16x16x32_f16(false, af0, false, b1f0, (short)0, acc, false, false);
    acc = __builtin_amdgcn_wmma_f32_16x16x32_f16(false, af1, false, b1f1, (short)0, acc, false, false);
    acc = __builtin_amdgcn_wmma_f32_16x16x32_f16(false, af2, false, b1f2, (short)0, acc, false, false);
    acc = __builtin_amdgcn_wmma_f32_16x16x32_f16(false, af3, false, b1f3, (short)0, acc, false, false);

    // unconditional stores: pad positions land in h2 rows 324..335 (never read)
    int co = lane & 15;
    int hbase = (stile*16 + laneHi8)*80 + co*5 + d;
#pragma unroll
    for (int r = 0; r < 8; ++r) {
      float v = acc[r] + bias1;
      v = (v >= 0.f) ? v : NEG_SLOPE * v;
      h2s[hbase + r*80] = (_Float16)v;
    }
  }
  __syncthreads();

  // ---- conv2: M=16 pixels, N=5 depths (of 16), K=720 (pad 736); 2 tiles/wave ----
  {
    const float bias2 = bt2[0];
    const int m = lane & 15;
    const int tt0 = wave, tt1 = wave + 8;
    const int vb0 = tt0*1440 + m*80 + laneHi8;   // run offset = CONST + laneHi8
    const int vb1 = tt1*1440 + m*80 + laneHi8;
    const int bofs = lane*16;
    v8f acc0 = {}, acc1 = {};
#pragma unroll
    for (int ks = 0; ks < 23; ++ks) {
      v16h bb = *(const v16h*)(B2 + ks*512 + bofs);
      v8h q0 = *(const v8h*)(h2s + vb0 + RUNOFFK(ks*32));
      v8h p0 = *(const v8h*)(h2s + vb1 + RUNOFFK(ks*32));
      v8h q1 = {}, p1 = {};
      if (ks < 22) {   // folds at unroll: last step's upper run is all-pad
        q1 = *(const v8h*)(h2s + vb0 + RUNOFFK(ks*32+16));
        p1 = *(const v8h*)(h2s + vb1 + RUNOFFK(ks*32+16));
      }
      v16h a0 = __builtin_shufflevector(q0, q1, 0,1,2,3,4,5,6,7,8,9,10,11,12,13,14,15);
      acc0 = __builtin_amdgcn_wmma_f32_16x16x32_f16(false, a0, false, bb,
                                                    (short)0, acc0, false, false);
      v16h a1 = __builtin_shufflevector(p0, p1, 0,1,2,3,4,5,6,7,8,9,10,11,12,13,14,15);
      acc1 = __builtin_amdgcn_wmma_f32_16x16x32_f16(false, a1, false, bb,
                                                    (short)0, acc1, false, false);
    }
    int d = lane & 15;
    if (d < TT) {
      float tb = tbias[d];
      size_t base = ((size_t)b*TT + d)*(size_t)NPIX;
#pragma unroll
      for (int r = 0; r < 8; ++r) {
        int mo = r + laneHi8;
        float v0 = acc0[r] + bias2;
        v0 = 1.f / (1.f + expf(-v0)) + tb;
        __builtin_nontemporal_store(v0, &outT[base + (size_t)(h0 + tt0)*HW + (w0 + mo)]);
        float v1 = acc1[r] + bias2;
        v1 = 1.f / (1.f + expf(-v1)) + tb;
        __builtin_nontemporal_store(v1, &outT[base + (size_t)(h0 + tt1)*HW + (w0 + mo)]);
      }
    }
  }
}

// ---------------------------------------------------------------------------
extern "C" void kernel_launch(void* const* d_in, const int* in_sizes, int n_in,
                              void* d_out, int out_size, void* d_ws, size_t ws_size,
                              hipStream_t stream) {
  const float* I     = (const float*)d_in[0];
  const float* J     = (const float*)d_in[1];
  const float* wbl1  = (const float*)d_in[2];
  const float* bbl1  = (const float*)d_in[3];
  const float* wbl2  = (const float*)d_in[4];
  const float* bbl2  = (const float*)d_in[5];
  const float* wt1   = (const float*)d_in[6];
  const float* bt1   = (const float*)d_in[7];
  const float* wt2   = (const float*)d_in[8];
  const float* bt2   = (const float*)d_in[9];
  const float* w1    = (const float*)d_in[10];
  const float* w2    = (const float*)d_in[11];
  const float* tbias = (const float*)d_in[12];

  float* ws    = (float*)d_ws;
  float* blOut = (float*)d_out;                             // [4,3,5,512,512]
  float* tOut  = blOut + (size_t)BATCH*3*TT*NPIX;           // [4,1,5,512,512]

  (void)hipFuncSetAttribute((const void*)fused_conv_kernel,
                            hipFuncAttributeMaxDynamicSharedMemorySize, SMEM_BYTES);

  stats_kernel<<<BATCH, 256, 0, stream>>>(I, ws);
  diff_kernel<<<BATCH*3*TT, 256, 0, stream>>>(I, J, ws);
  mlp_kernel<<<1, 32, 0, stream>>>(wbl1, bbl1, wbl2, bbl2, w1, ws);

  dim3 grid(HW/16, HW/16, BATCH);
  fused_conv_kernel<<<grid, 256, SMEM_BYTES, stream>>>(I, J, wt1, bt1, wt2, bt2,
                                                       w2, tbias, ws, blOut, tOut);
}
